// ContrastiveSSL_81475529605498
// MI455X (gfx1250) — matrix operand order repaired and checked by reading, not compiled, not run on hardware
//
#include <hip/hip_runtime.h>
#include <hip/hip_bf16.h>
#include <math.h>

// Problem constants (from reference setup_inputs)
#define NB   8
#define CC   256
#define LLEN 512
#define KK   100
#define NL   (NB * LLEN)        // 4096 rows
#define NPASS ((KK + 15) / 16)  // 7 WMMA passes of 16 negatives
#define TEMP_INV 2.0f           // 1 / 0.5
#define EPSN 1e-8f

typedef __attribute__((ext_vector_type(16))) __bf16 v16bf;
typedef __attribute__((ext_vector_type(8)))  __bf16 v8bf;
typedef __attribute__((ext_vector_type(8)))  float  v8f;

// ---------- wave32 reduction helpers ----------
__device__ __forceinline__ float wave_reduce_add(float v) {
  #pragma unroll
  for (int off = 16; off > 0; off >>= 1) v += __shfl_xor(v, off, 32);
  return v;
}
__device__ __forceinline__ float wave_reduce_max(float v) {
  #pragma unroll
  for (int off = 16; off > 0; off >>= 1) v = fmaxf(v, __shfl_xor(v, off, 32));
  return v;
}
__device__ __forceinline__ unsigned wave_reduce_xor(unsigned v) {
  #pragma unroll
  for (int off = 16; off > 0; off >>= 1) v ^= (unsigned)__shfl_xor((int)v, off, 32);
  return v;
}

// ---------- Kernel 1: tiled transpose (C,L)->(L,C) per batch ----------
// src: (NB, CC, srcW) f32; dst row (n*LLEN + l) gets src column (l + colOff).
__global__ void __launch_bounds__(256)
xpose_kernel(const float* __restrict__ src, float* __restrict__ dst,
             int srcW, int colOff) {
  __shared__ float tile[32][33];      // +1 pad: no LDS bank conflicts
  const int n  = blockIdx.z;
  const int l0 = blockIdx.x * 32;
  const int c0 = blockIdx.y * 32;
  const int tx = threadIdx.x, ty = threadIdx.y;
  #pragma unroll
  for (int i = 0; i < 4; ++i) {
    const int ch  = c0 + ty + i * 8;
    const int col = l0 + tx + colOff;          // coalesced along tx
    tile[ty + i * 8][tx] = src[((size_t)n * CC + ch) * srcW + col];
  }
  __syncthreads();
  #pragma unroll
  for (int i = 0; i < 4; ++i) {
    const int l  = l0 + ty + i * 8;
    const int ch = c0 + tx;                    // coalesced along tx
    dst[((size_t)n * LLEN + l) * CC + ch] = tile[tx][ty + i * 8];
  }
}

// ---------- Kernel 2: per-row inv-norm, bf16 copy, equality hash ----------
// One wave per row; rows 0..NL-1 = zs, NL..2NL-1 = cs.
__global__ void __launch_bounds__(32)
rowprep_kernel(const float* __restrict__ zs, const float* __restrict__ cs,
               __bf16* __restrict__ zsb, __bf16* __restrict__ csb,
               float* __restrict__ zinv, float* __restrict__ cinv,
               unsigned* __restrict__ zhash, unsigned* __restrict__ chash) {
  const int r    = blockIdx.x;
  const int lane = threadIdx.x;
  const bool isC = (r >= NL);
  const int row  = isC ? (r - NL) : r;
  const float* srow = (isC ? cs : zs) + (size_t)row * CC;
  const float4* s4  = (const float4*)srow;
  const float4 a = s4[lane * 2];       // elems lane*8 .. lane*8+3 (fully coalesced)
  const float4 b = s4[lane * 2 + 1];   // elems lane*8+4 .. lane*8+7

  float ss = a.x*a.x + a.y*a.y + a.z*a.z + a.w*a.w
           + b.x*b.x + b.y*b.y + b.z*b.z + b.w*b.w;
  unsigned h = 2166136261u;
  h = (h * 16777619u) ^ __float_as_uint(a.x);
  h = (h * 16777619u) ^ __float_as_uint(a.y);
  h = (h * 16777619u) ^ __float_as_uint(a.z);
  h = (h * 16777619u) ^ __float_as_uint(a.w);
  h = (h * 16777619u) ^ __float_as_uint(b.x);
  h = (h * 16777619u) ^ __float_as_uint(b.y);
  h = (h * 16777619u) ^ __float_as_uint(b.z);
  h = (h * 16777619u) ^ __float_as_uint(b.w);

  const float tot  = wave_reduce_add(ss);
  const unsigned hh = wave_reduce_xor(h * (2u * (unsigned)lane + 1u));
  const float inv  = 1.0f / fmaxf(sqrtf(tot), EPSN);

  __bf16* drow = (isC ? csb : zsb) + (size_t)row * CC;
  v8bf o;
  o[0] = (__bf16)a.x; o[1] = (__bf16)a.y; o[2] = (__bf16)a.z; o[3] = (__bf16)a.w;
  o[4] = (__bf16)b.x; o[5] = (__bf16)b.y; o[6] = (__bf16)b.z; o[7] = (__bf16)b.w;
  *(v8bf*)(drow + lane * 8) = o;                // contiguous 512B per wave

  if (lane == 0) {
    if (isC) { cinv[row] = inv; chash[row] = hh; }
    else     { zinv[row] = inv; zhash[row] = hh; }
  }
}

// Batched gather of one negative tile (16 rows x 32B chunks) into registers.
// Emits a 16 x global_load_b128 clause -> latency amortized, WMMAs run
// back-to-back off the register file.
__device__ __forceinline__ void load_Btile(const __bf16* __restrict__ zsb,
                                           int idx, int half, v16bf* B) {
  const __bf16* brow = zsb + (size_t)idx * CC + half * 16;
  #pragma unroll
  for (int t = 0; t < 8; ++t) B[t] = *(const v16bf*)(brow + t * 32);
}

// ---------- Kernel 3: WMMA logits + online logsumexp ----------
// 8 waves per block, one wave per position p. 16 negatives per WMMA pass,
// C=256 chunked by 32 -> 8 x v_wmma_f32_16x16x32_bf16 per pass.
// Fully unrolled 7-pass loop with a two-deep B ping-pong so pass k+1's
// gathers are in flight while pass k's WMMAs and reductions execute.
__global__ void __launch_bounds__(256)
loss_kernel(const float* __restrict__ zs,  const float* __restrict__ cs,
            const __bf16* __restrict__ zsb, const __bf16* __restrict__ csb,
            const float* __restrict__ zinv, const float* __restrict__ cinv,
            const unsigned* __restrict__ zhash, const unsigned* __restrict__ chash,
            const int* __restrict__ neg_inds, float* __restrict__ accum) {
  const int lane = threadIdx.x & 31;
  const int wv   = threadIdx.x >> 5;
  const int p    = blockIdx.x * 8 + wv;          // exact cover: no divergence
  const int half = lane >> 4;                    // K-half for WMMA layouts
  const int col  = lane & 15;                    // output column / negative id

  // ---- prefetch ALL negative indices for this position (one load clause) ----
  int idxs[NPASS];
  #pragma unroll
  for (int pass = 0; pass < NPASS; ++pass) {
    const int j = pass * 16 + col;
    idxs[pass] = (lane < 16 && j < KK) ? neg_inds[(size_t)p * KK + j] : 0;
  }

  // ---- positive logit in f32 (full precision path) ----
  const float* crow = cs + (size_t)p * CC;
  const float* zrow = zs + (size_t)p * CC;
  const float4* c4 = (const float4*)crow;
  const float4* z4 = (const float4*)zrow;
  const float4 ca = c4[lane*2], cb4 = c4[lane*2+1];
  const float4 za = z4[lane*2], zb4 = z4[lane*2+1];
  float pd = ca.x*za.x + ca.y*za.y + ca.z*za.z + ca.w*za.w
           + cb4.x*zb4.x + cb4.y*zb4.y + cb4.z*zb4.z + cb4.w*zb4.w;
  const float posdot = wave_reduce_add(pd);
  const float ci  = cinv[p];
  const float pos = posdot * ci * zinv[p] * TEMP_INV;
  const unsigned chp = chash[p];

  // ---- hoist A operand (c row, bf16) — identical across all 7 passes ----
  // A 16x32 layout (contiguous, matching B): lanes 0-15 hold K 0-15;
  // lanes 16-31 hold K 16-31. Rows replicated (same c vector).
  const __bf16* arow = csb + (size_t)p * CC;
  v16bf Areg[8];
  #pragma unroll
  for (int t = 0; t < 8; ++t) {
    Areg[t] = *(const v16bf*)(arow + t * 32 + half * 16);
  }

  // ---- software-pipelined pass loop (compile-time unrolled ping-pong) ----
  v16bf Bbuf[2][8];
  int idx_cur = __shfl(idxs[0], col, 32);        // both halves need column idx
  load_Btile(zsb, idx_cur, half, Bbuf[0]);

  float m = pos, s = 1.0f;

  #pragma unroll
  for (int pass = 0; pass < NPASS; ++pass) {
    // issue next pass's 16-load gather clause BEFORE consuming current tile
    int idx_next = 0;
    if (pass + 1 < NPASS) {
      idx_next = __shfl(idxs[pass + 1], col, 32);
      load_Btile(zsb, idx_next, half, Bbuf[(pass + 1) & 1]);
    }

    const int j = pass * 16 + col;
    const bool valid = (j < KK);
    const int idx = idx_cur;

    v8f acc = {};
    #pragma unroll
    for (int t = 0; t < 8; ++t) {
      acc = __builtin_amdgcn_wmma_f32_16x16x32_bf16(
                false, Areg[t], false, Bbuf[pass & 1][t], (short)0, acc,
                false, false);
    }

    float logit = -INFINITY;
    bool hcand = false;
    if (lane < 16 && valid) {
      logit = acc[0] * ci * zinv[idx] * TEMP_INV;   // D row 0, col = lane
      hcand = (chp == zhash[idx]);
    }

    // exact neg_in_target check only on hash match (≈never taken)
    unsigned long long em = __ballot(hcand);
    while (em) {
      const int jb = __ffsll(em) - 1;
      em &= em - 1;
      const int bidx = __shfl(idx, jb, 32);
      const float* nrow = zs + (size_t)bidx * CC;
      bool eq = true;
      #pragma unroll
      for (int t = 0; t < 8; ++t)
        eq = eq && (crow[lane * 8 + t] == nrow[lane * 8 + t]);
      const bool alleq = __all(eq);
      if (alleq && lane == jb) logit = -INFINITY;
    }

    // fold 16 new logits into running (m, s); invalid/upper lanes give exp->0
    const float pm = wave_reduce_max(logit);
    const float nm = fmaxf(m, pm);
    const float e  = __expf(logit - nm);
    const float ps = wave_reduce_add(e);
    s = s * __expf(m - nm) + ps;
    m = nm;

    idx_cur = idx_next;
  }

  if (lane == 0) atomicAdd(accum, __logf(s) + m - pos);
}

// ---------- Kernel 4: finalize mean ----------
__global__ void finalize_kernel(const float* __restrict__ accum,
                                float* __restrict__ out) {
  out[0] = accum[0] * (1.0f / (float)NL);
}

extern "C" void kernel_launch(void* const* d_in, const int* in_sizes, int n_in,
                              void* d_out, int out_size, void* d_ws, size_t ws_size,
                              hipStream_t stream) {
  const float* z   = (const float*)d_in[0];   // (8, 256, 512)  f32
  const float* c   = (const float*)d_in[1];   // (8, 256, 513)  f32
  const int*   neg = (const int*)d_in[2];     // (8, 512, 100)  i32

  // Workspace layout (~12.1 MB)
  char* ws = (char*)d_ws;
  float*    zs    = (float*)(ws);                               // 4 MiB
  float*    cs    = (float*)(ws + (4u  << 20));                 // 4 MiB
  __bf16*   zsb   = (__bf16*)(ws + (8u  << 20));                // 2 MiB
  __bf16*   csb   = (__bf16*)(ws + (10u << 20));                // 2 MiB
  float*    zinv  = (float*)(ws + (12u << 20));                 // 16 KiB
  float*    cinv  = (float*)(ws + (12u << 20) + (16u << 10));
  unsigned* zhash = (unsigned*)(ws + (12u << 20) + (32u << 10));
  unsigned* chash = (unsigned*)(ws + (12u << 20) + (48u << 10));
  float*    accum = (float*)(ws + (12u << 20) + (64u << 10));

  const dim3 tb(32, 8);
  // z: (C,L) -> row-major (N*L, C)
  xpose_kernel<<<dim3(LLEN / 32, CC / 32, NB), tb, 0, stream>>>(z, zs, LLEN, 0);
  // c: drop start token -> column l+1 becomes row l
  xpose_kernel<<<dim3(LLEN / 32, CC / 32, NB), tb, 0, stream>>>(c, cs, LLEN + 1, 1);

  rowprep_kernel<<<2 * NL, 32, 0, stream>>>(zs, cs, zsb, csb,
                                            zinv, cinv, zhash, chash);

  hipMemsetAsync(accum, 0, sizeof(float), stream);

  loss_kernel<<<NL / 8, 256, 0, stream>>>(zs, cs, zsb, csb, zinv, cinv,
                                          zhash, chash, neg, accum);

  finalize_kernel<<<1, 1, 0, stream>>>(accum, (float*)d_out);
}
